// ApproxLTCLayer_38594576122136
// MI455X (gfx1250) — compile-verified, matrix operand order checked by reading, not run on hardware
//
#include <hip/hip_runtime.h>

typedef __attribute__((ext_vector_type(16))) _Float16 v16h;
typedef __attribute__((ext_vector_type(8)))  float    v8f;

// ---------------------------------------------------------------------------
// Kernel 1: sumA[u] = sum_d A[u,d]  (U=64, D=16) via v_wmma_f32_16x16x32_f16.
// f32 accuracy kept with a hi/lo f16 split: A = hi + lo, hi occupies K=0..15,
// lo occupies K=16..31, B = all-ones(32x16)  =>  C[m,*] = sum_d A[m,d] exactly
// to f32 rounding. One WMMA per 16-row tile of A; 4 waves total.
// A 16-bit A-matrix layout (16x32): lane<16 holds K=0..7 (v[0..7]) and
// K=16..23 (v[8..15]); lane>=16 holds K=8..15 and K=24..31.
// ---------------------------------------------------------------------------
__global__ __launch_bounds__(32) void sumA_wmma_kernel(
    const float* __restrict__ A, float* __restrict__ sumA)
{
  const int lane  = threadIdx.x;          // one wave per block
  const int utile = blockIdx.x;           // 0..3 -> rows 16*utile..+15
  const int row   = (lane & 15) + 16 * utile;
  const int dbase = (lane < 16) ? 0 : 8;

  v16h a, b;
#pragma unroll
  for (int i = 0; i < 8; ++i) {
    const float av = A[row * 16 + dbase + i];
    const _Float16 hi = (_Float16)av;
    const _Float16 lo = (_Float16)(av - (float)hi);
    a[i]     = hi;               // K = dbase+i
    a[i + 8] = lo;               // K = 16 + dbase + i
    b[i]     = (_Float16)1.0f;   // all-ones B: layout-independent
    b[i + 8] = (_Float16)1.0f;
  }
  v8f c = {};
  c = __builtin_amdgcn_wmma_f32_16x16x32_f16(
      /*neg_a=*/false, a, /*neg_b=*/false, b,
      /*c_mod=*/(short)0, c, /*reuse_a=*/false, /*reuse_b=*/false);

  // C 16x16 f32 layout: VGPR r, lanes 0-15 -> M=r; lanes 16-31 -> M=r+8.
  // All 16 columns are identical (B was all-ones); extract from lanes 0 and 16.
  if (lane == 0) {
#pragma unroll
    for (int r = 0; r < 8; ++r) sumA[16 * utile + r] = c[r];
  } else if (lane == 16) {
#pragma unroll
    for (int r = 0; r < 8; ++r) sumA[16 * utile + 8 + r] = c[r];
  }
}

// ---------------------------------------------------------------------------
// Async x-tile staging (wave 0 only; 64 floats = 32 lanes x 8 B per issue).
// Inline CDNA5 asm: the clang builtin's pointer params live in the unspellable
// cuda_device/cuda_shared language address spaces, so we emit the instruction
// directly.  VDST = per-lane LDS byte address (low 32 bits of the flat shared
// pointer == LDS offset per ISA 10.2), VADDR = per-lane 64-bit global address.
// Tracked with ASYNCcnt; waited with s_wait_asynccnt 0.
// ---------------------------------------------------------------------------
__device__ __forceinline__ void copy_x_async(const float* __restrict__ src,
                                             float* lds, int lane)
{
  const unsigned lds_addr =
      (unsigned)(unsigned long long)(lds + 2 * lane);   // LDS byte offset
  const float* gaddr = src + 2 * lane;
  asm volatile("global_load_async_to_lds_b64 %0, %1, off"
               :
               : "v"(lds_addr), "v"(gaddr)
               : "memory");
}

__device__ __forceinline__ void wait_async_zero()
{
  asm volatile("s_wait_asynccnt 0" ::: "memory");
}

// ---------------------------------------------------------------------------
// Kernel 2: main LTC evaluation.
// Grid: 1024 blocks x 256 threads. Each block owns 32 pixels of one batch b
// (128 blocks per b, so t = b is block-uniform), processed as 8 groups of
// 4 pixels. Thread t -> (u = t&63, p_local = t>>6); per-wave u is contiguous
// (conflict-free LDS param reads, broadcast x reads, coalesced stores).
// x tiles are double-buffered into LDS with async copies overlapping compute.
// ---------------------------------------------------------------------------
__global__ __launch_bounds__(256) void ltc_main_kernel(
    const float* __restrict__ x,     // (8,4096,16)
    const float* __restrict__ A,     // (64,16)
    const float* __restrict__ sigma, // (64,16)
    const float* __restrict__ mu,    // (64,16)
    const float* __restrict__ x0,    // (64,)
    const float* __restrict__ sumA,  // (64,) from WMMA kernel
    float* __restrict__ out)         // (8,4096,64)
{
  __shared__ float s_sig[16 * 64];   // [d*64 + u] : sigma
  __shared__ float s_smu[16 * 64];   // [d*64 + u] : sigma*mu
  __shared__ float s_w  [16 * 64];   // [d*64 + u] : x0[u]-A[u,d]
  __shared__ float s_sumA[64];
  __shared__ float s_x[2][64];       // double-buffered 4-pixel x tiles

  const int tid = threadIdx.x;

  // Stage transposed parameter tables (coalesced global reads).
  for (int i = tid; i < 1024; i += 256) {
    const int u = i >> 4, d = i & 15;
    const int ld = d * 64 + u;
    const float sg = sigma[i];
    s_sig[ld] = sg;
    s_smu[ld] = sg * mu[i];
    s_w[ld]   = x0[u] - A[i];
  }
  if (tid < 64) s_sumA[tid] = sumA[tid];

  const int   gp_base = blockIdx.x * 32;          // first pixel of block
  const float tb   = (float)(blockIdx.x >> 7);    // t = b (block-uniform)
  const float NL2E = -1.44269504f;                // -log2(e)
  const float tn   = tb * NL2E;                   // -t*log2(e)
  const float tbo  = 0.1f * tn;                   // -0.1*t*log2(e)

  const int u       = tid & 63;
  const int p_local = tid >> 6;

  // Prologue: start async copy of group 0.
  if (tid < 32) copy_x_async(x + gp_base * 16, &s_x[0][0], tid);

  for (int g = 0; g < 8; ++g) {
    if (tid < 32) wait_async_zero();   // copy g landed in LDS
    __syncthreads();                   // visible to all; prev compute retired
    if ((tid < 32) && (g + 1 < 8))     // overlap copy g+1 with compute g
      copy_x_async(x + (gp_base + (g + 1) * 4) * 16, &s_x[(g + 1) & 1][0], tid);

    const float* sx = &s_x[g & 1][p_local * 16];
    float acc = s_sumA[u];             // + sum_d A[u,d] term
#pragma unroll
    for (int d = 0; d < 16; ++d) {
      const int   ld = d * 64 + u;
      const float xd = sx[d];                                  // LDS broadcast
      const float z  = fmaf(s_sig[ld], xd, -s_smu[ld]);        // sigma*(x-mu)
      const float ez = __builtin_amdgcn_exp2f(z * NL2E);       // exp(-z)
      const float fp = __builtin_amdgcn_rcpf(1.0f + ez);       // sigmoid(z)
      const float fn = ez * fp;                                // sigmoid(-z)
      const float e  = __builtin_amdgcn_exp2f(fmaf(fp, tn, tbo)); // exp(-(.1+fp)t)
      acc = fmaf(s_w[ld] * fn, e, acc);
    }
    out[(gp_base + (g << 2) + p_local) * 64 + u] = acc;        // coalesced
  }
}

// ---------------------------------------------------------------------------
extern "C" void kernel_launch(void* const* d_in, const int* in_sizes, int n_in,
                              void* d_out, int out_size, void* d_ws, size_t ws_size,
                              hipStream_t stream)
{
  const float* x     = (const float*)d_in[0]; // inputs (8,64,64,16)
  const float* A     = (const float*)d_in[1]; // (64,16)
  const float* sigma = (const float*)d_in[2]; // (64,16)
  const float* mu    = (const float*)d_in[3]; // (64,16)
  const float* x0    = (const float*)d_in[4]; // (64,)
  float* out  = (float*)d_out;                // (8,4096,64)
  float* sumA = (float*)d_ws;                 // 64 floats of scratch

  sumA_wmma_kernel<<<4, 32, 0, stream>>>(A, sumA);
  ltc_main_kernel<<<1024, 256, 0, stream>>>(x, A, sigma, mu, x0, sumA, out);
}
